// SSL_BACKEND_BAM_2267742732705
// MI455X (gfx1250) — matrix operand order repaired and checked by reading, not compiled, not run on hardware
//
#include <hip/hip_runtime.h>
#include <hip/hip_bf16.h>

#define B_      2
#define T_      256
#define IN_DIM  512
#define OUT_DIM 256
#define H_      8
#define KEXT    4608   // H_*IN_DIM + IN_DIM
#define KCN     144    // KEXT/32

typedef __attribute__((ext_vector_type(16))) __bf16 v16bf;
typedef __attribute__((ext_vector_type(2)))  __bf16 bf2;
typedef __attribute__((ext_vector_type(8)))  float  v8f;

struct U8x32 { uint4 lo, hi; };

static __device__ __forceinline__ unsigned short f2bfu(float f) {
    unsigned u = __builtin_bit_cast(unsigned, f);
    unsigned r = u + 0x7FFFu + ((u >> 16) & 1u);   // round-to-nearest-even
    return (unsigned short)(r >> 16);
}
static __device__ __forceinline__ v16bf make_v16(uint4 a, uint4 b) {
    U8x32 t{a, b};
    return __builtin_bit_cast(v16bf, t);
}
static __device__ __forceinline__ unsigned pack2(float a, float b) {
    return (unsigned)f2bfu(a) | ((unsigned)f2bfu(b) << 16);
}
// packed bf16 * bf16 (lowers to v_pk_mul_bf16 when available)
static __device__ __forceinline__ unsigned pkmul(unsigned q, bf2 s) {
    bf2 v = __builtin_bit_cast(bf2, q);
    v = v * s;
    return __builtin_bit_cast(unsigned, v);
}
static __device__ __forceinline__ float fast_tanh(float x) {
#if defined(__HIP_DEVICE_COMPILE__)
  #if __has_builtin(__builtin_amdgcn_tanhf)
    return __builtin_amdgcn_tanhf(x);
  #elif __has_builtin(__builtin_amdgcn_tanh_f32)
    return __builtin_amdgcn_tanh_f32(x);
  #else
    float e = __expf(2.0f * x);
    return 1.0f - 2.0f / (e + 1.0f);
  #endif
#else
    return tanhf(x);
#endif
}

// ---------------- workspace layout (bytes) ----------------
#define OFF_WSWZ 0          // att_proj_w bf16 A-fragments:    262144
#define OFF_XT   262144     // x transposed [b][d][k] bf16:    524288
#define OFF_WEXT 786432     // [paw;pnw] bf16 B-fragments:     2359296
#define OFF_AW2  3145728    // att_weight bf16 B-fragments:    8192
#define OFF_S    3153920    // boundary prefix sums:           2048
#define OFF_X1E  3155968    // [512][4608] bf16 extended x1:   4718592
#define OFF_Y    7874560    // pre-BN output (512,256) f32:    524288

// ---------------- prep kernels ----------------

// att_proj_w (256x512) -> bf16 WMMA A-fragment order
// flat = ((o_tile*16 + d_chunk)*32 + lane)*16 + e
__global__ void prep_wswz(const float* __restrict__ W, unsigned short* __restrict__ out) {
    int t = blockIdx.x * blockDim.x + threadIdx.x;           // 131072
    int e = t & 15, lane = (t >> 4) & 31, dc = (t >> 9) & 15, ot = t >> 13;
    int o = ot * 16 + (lane & 15);
    int K = (e < 8 ? e : e + 8) + ((lane & 16) ? 8 : 0);
    out[t] = f2bfu(W[o * IN_DIM + dc * 32 + K]);
}

// x (B,T,IN) -> Xt[b][d][k] bf16
__global__ void prep_xt(const float* __restrict__ x, unsigned short* __restrict__ xt) {
    int t = blockIdx.x * blockDim.x + threadIdx.x;           // 262144
    int k = t & 255, d = (t >> 8) & 511, b = t >> 17;
    xt[t] = f2bfu(x[(b * T_ + k) * IN_DIM + d]);
}

// [proj_att_w ; proj_noatt_w] -> bf16 WMMA B-fragment order over K=4608
__global__ void prep_wext(const float* __restrict__ paw, const float* __restrict__ pnw,
                          unsigned short* __restrict__ out) {
    int t = blockIdx.x * blockDim.x + threadIdx.x;           // 1179648
    int e = t & 15, lane = (t >> 4) & 31, nt = (t >> 9) & 15, kc = t >> 13;
    int o = nt * 16 + (lane & 15);
    int i = kc * 32 + (e < 8 ? e : e + 8) + ((lane & 16) ? 8 : 0);
    float v = (i < H_ * IN_DIM) ? paw[o * (H_ * IN_DIM) + i]
                                : pnw[o * IN_DIM + (i - H_ * IN_DIM)];
    out[t] = f2bfu(v);
}

// att_weight (256x8) -> bf16 B-fragments, N padded to 16
__global__ void prep_aw2(const float* __restrict__ awt, unsigned short* __restrict__ out) {
    int t = blockIdx.x * blockDim.x + threadIdx.x;           // 4096
    int e = t & 15, lane = (t >> 4) & 31, og = t >> 9;
    int n = lane & 15;
    int o = og * 32 + (e < 8 ? e : e + 8) + ((lane & 16) ? 8 : 0);
    out[t] = (n < 8) ? f2bfu(awt[o * 8 + n]) : (unsigned short)0;
}

// tail of extended x1 rows: X1e[row][4096 + d] = bf16(x[row][d])
__global__ void prep_x1tail(const float* __restrict__ x, unsigned short* __restrict__ x1e) {
    int t = blockIdx.x * blockDim.x + threadIdx.x;           // 262144
    int d = t & 511, row = t >> 9;
    x1e[(size_t)row * KEXT + H_ * IN_DIM + d] = f2bfu(x[(size_t)row * IN_DIM + d]);
}

// inclusive prefix sums of boundary per batch
__global__ void prep_scan(const int* __restrict__ bnd, int* __restrict__ S) {
    int b = threadIdx.x;
    if (b < B_) {
        int acc = 0;
        for (int t = 0; t < T_; ++t) { acc += bnd[b * T_ + t]; S[b * T_ + t] = acc; }
    }
}

// ---------------- main fused kernel: one workgroup per (b,j) ----------------
__global__ __launch_bounds__(256)
void bam_main(const float* __restrict__ x,
              const unsigned short* __restrict__ wswz,
              const unsigned short* __restrict__ xt,
              const float* __restrict__ apb,
              const unsigned short* __restrict__ aw2,
              const int*   __restrict__ Sarr,
              unsigned short* __restrict__ x1e) {
    const int bid = blockIdx.x;
    const int b = bid / T_;
    const int j = bid % T_;
    const int tid  = threadIdx.x;
    const int w    = tid >> 5;
    const int lane = tid & 31;
    const int mofs = (lane & 16) ? 8 : 0;

    __shared__ float s_xj[IN_DIM];                      // 2 KB
    __shared__ float s_apb[OUT_DIM];                    // 1 KB
    __shared__ float s_att[T_ * H_];                    // 8 KB
    __shared__ float s_attp[8 * 16 * H_];               // 4 KB per-wave partials
    __shared__ __align__(32) float s_dyn[IN_DIM * H_];  // 16 KB (bslab, then x1buf)
    unsigned short* s_bslab = (unsigned short*)s_dyn;

    // phase 0: stage xj / bias
    s_xj[tid]       = x[((size_t)b * T_ + j) * IN_DIM + tid];
    s_xj[tid + 256] = x[((size_t)b * T_ + j) * IN_DIM + tid + 256];
    s_apb[tid] = apb[tid];

    // att_weight B-fragment for this wave's o-range [32w, 32w+32) — lives in regs
    const uint4* awp = (const uint4*)(aw2 + (size_t)((w * 32 + lane) << 4));
    const v16bf awf = make_v16(awp[0], awp[1]);
    __syncthreads();

    // ---- phase 1: a = tanh(W (xj.xk) + b); contract with att_weight via WMMA ----
    #pragma unroll 1
    for (int kb = 0; kb < 16; ++kb) {
        // shared B-slab: bslab[d][kk] = bf16(xj[d]) * Xt[b][d][kb*16+kk]  (512x16)
        #pragma unroll
        for (int rr = 0; rr < 2; ++rr) {
            int d = tid + rr * 256;
            const uint4* xp = (const uint4*)(xt + ((size_t)(b * IN_DIM + d)) * T_ + kb * 16);
            uint4 q0 = xp[0], q1 = xp[1];
            unsigned short us = f2bfu(s_xj[d]);
            unsigned up = (unsigned)us | ((unsigned)us << 16);
            bf2 sc = __builtin_bit_cast(bf2, up);
            uint4 r0, r1;
            r0.x = pkmul(q0.x, sc); r0.y = pkmul(q0.y, sc);
            r0.z = pkmul(q0.z, sc); r0.w = pkmul(q0.w, sc);
            r1.x = pkmul(q1.x, sc); r1.y = pkmul(q1.y, sc);
            r1.z = pkmul(q1.z, sc); r1.w = pkmul(q1.w, sc);
            uint4* dst = (uint4*)(s_bslab + d * 16);
            dst[0] = r0; dst[1] = r1;
        }
        __syncthreads();

        // two consecutive o-tiles per wave, shared B fragment
        v8f c0 = {0.f,0.f,0.f,0.f,0.f,0.f,0.f,0.f};
        v8f c1 = {0.f,0.f,0.f,0.f,0.f,0.f,0.f,0.f};
        const int ot0 = w * 2;
        #pragma unroll 4
        for (int dc = 0; dc < 16; ++dc) {
            const uint4* a0p = (const uint4*)(wswz + (size_t)((((ot0    ) * 16 + dc) * 32 + lane) << 4));
            const uint4* a1p = (const uint4*)(wswz + (size_t)((((ot0 + 1) * 16 + dc) * 32 + lane) << 4));
            v16bf af0 = make_v16(a0p[0], a0p[1]);
            v16bf af1 = make_v16(a1p[0], a1p[1]);
            const unsigned short* bp = s_bslab + (dc * 32 + mofs) * 16 + (lane & 15);
            v16bf bf;
            #pragma unroll
            for (int s = 0; s < 16; ++s) bf[s] = __builtin_bit_cast(__bf16, bp[(s < 8 ? s : s + 8) * 16]);
            c0 = __builtin_amdgcn_wmma_f32_16x16x32_bf16(false, af0, false, bf, (short)0, c0, false, false);
            c1 = __builtin_amdgcn_wmma_f32_16x16x32_bf16(false, af1, false, bf, (short)0, c1, false, false);
        }
        // tanh(+bias), pack the two tiles into one A-fragment with K = o (32 values)
        float t0[8], t1[8];
        #pragma unroll
        for (int r = 0; r < 8; ++r) {
            t0[r] = fast_tanh(c0[r] + s_apb[ ot0      * 16 + r + mofs]);
            t1[r] = fast_tanh(c1[r] + s_apb[(ot0 + 1) * 16 + r + mofs]);
        }
        uint4 lo, hi;
        lo.x = pack2(t0[0], t0[1]); lo.y = pack2(t0[2], t0[3]);
        lo.z = pack2(t0[4], t0[5]); lo.w = pack2(t0[6], t0[7]);
        hi.x = pack2(t1[0], t1[1]); hi.y = pack2(t1[2], t1[3]);
        hi.z = pack2(t1[4], t1[5]); hi.w = pack2(t1[6], t1[7]);
        v16bf a2 = make_v16(lo, hi);
        v8f c2 = {0.f,0.f,0.f,0.f,0.f,0.f,0.f,0.f};
        c2 = __builtin_amdgcn_wmma_f32_16x16x32_bf16(false, a2, false, awf, (short)0, c2, false, false);
        // c2: M = k within block, N = h; store per-wave partial (deterministic)
        const int h = lane & 15;
        if (h < 8) {
            #pragma unroll
            for (int r = 0; r < 8; ++r)
                s_attp[(w * 16 + r + mofs) * H_ + h] = c2[r];
        }
        __syncthreads();
        // fixed-order cross-wave reduction of the 16x8 logit block
        if (tid < 128) {
            int kl = tid >> 3, hh = tid & 7;
            float s = 0.0f;
            #pragma unroll
            for (int w2 = 0; w2 < 8; ++w2) s += s_attp[(w2 * 16 + kl) * H_ + hh];
            s_att[(kb * 16 + kl) * H_ + hh] = s;
        }
        __syncthreads();
    }

    // ---- phase 2: boundary mask + softmax over k (wave w handles head h=w) ----
    {
        const int* Sb = Sarr + b * T_;
        float vals[8];
        float mx = -1e30f;
        #pragma unroll
        for (int i = 0; i < 8; ++i) {
            int k = i * 32 + lane;
            float v = s_att[k * 8 + w];
            int m = 1;
            if (k != j) {
                int lo = k < j ? k : j, hib = k < j ? j : k;
                int cnt = Sb[hib] - (lo ? Sb[lo - 1] : 0);
                m = (cnt == 0);
            }
            v = m ? v : 0.0f;
            vals[i] = v;
            mx = fmaxf(mx, v);
        }
        #pragma unroll
        for (int off = 16; off; off >>= 1) mx = fmaxf(mx, __shfl_xor(mx, off, 32));
        float sum = 0.0f;
        #pragma unroll
        for (int i = 0; i < 8; ++i) { vals[i] = __expf(vals[i] - mx); sum += vals[i]; }
        #pragma unroll
        for (int off = 16; off; off >>= 1) sum += __shfl_xor(sum, off, 32);
        float inv = 1.0f / sum;
        #pragma unroll
        for (int i = 0; i < 8; ++i) s_att[(i * 32 + lane) * 8 + w] = vals[i] * inv;
    }
    __syncthreads();

    // ---- phase 3: x1[d,h] = sum_k att[k,h] * x[b,k,d] via WMMA ----
    {
        float x1loc[4][8];
        #pragma unroll 1
        for (int q = 0; q < 4; ++q) {
            const int dt = w * 4 + q;
            v8f c = {0.f,0.f,0.f,0.f,0.f,0.f,0.f,0.f};
            const int d = dt * 16 + (lane & 15);
            const int kofs = (lane & 16) ? 8 : 0;
            #pragma unroll 2
            for (int kc = 0; kc < 8; ++kc) {
                const unsigned short* xr = xt + ((size_t)(b * IN_DIM + d)) * T_ + kc * 32 + kofs;
                uint4 q0 = *(const uint4*)xr;
                uint4 q1 = *(const uint4*)(xr + 16);
                v16bf af = make_v16(q0, q1);
                const int hc = lane & 15;
                v16bf bfr;
                #pragma unroll
                for (int s = 0; s < 16; ++s) {
                    int k = kc * 32 + (s < 8 ? s : s + 8) + kofs;
                    bfr[s] = __builtin_bit_cast(__bf16, (hc < 8) ? f2bfu(s_att[k * 8 + hc]) : (unsigned short)0);
                }
                c = __builtin_amdgcn_wmma_f32_16x16x32_bf16(false, af, false, bfr, (short)0, c, false, false);
            }
            #pragma unroll
            for (int r = 0; r < 8; ++r) x1loc[q][r] = c[r];
        }
        __syncthreads();
        const int h = lane & 15;
        if (h < 8) {
            #pragma unroll
            for (int q = 0; q < 4; ++q) {
                const int dt = w * 4 + q;
                #pragma unroll
                for (int r = 0; r < 8; ++r)
                    s_dyn[(dt * 16 + r + mofs) * 8 + h] = x1loc[q][r];
            }
        }
    }
    __syncthreads();

    // ---- phase 4: pack x1 row to bf16 into X1e[row][0..4095] (coalesced dwords) ----
    {
        unsigned* dst = (unsigned*)(x1e + (size_t)bid * KEXT);
        #pragma unroll
        for (int ii = 0; ii < 8; ++ii) {
            int i = tid + ii * 256;                      // 2048 dwords
            dst[i] = pack2(s_dyn[2 * i], s_dyn[2 * i + 1]);
        }
    }
}

// ---------------- GEMM2: Y = X1e (512x4608) @ Wext^T + biases, via WMMA ----------------
__global__ __launch_bounds__(256)
void gemm2(const unsigned short* __restrict__ x1e,
           const unsigned short* __restrict__ wext,
           const float* __restrict__ pab,
           const float* __restrict__ pnb,
           float* __restrict__ Y) {
    const int mtile = blockIdx.x;          // 32 row tiles
    const int w     = threadIdx.x >> 5;
    const int lane  = threadIdx.x & 31;
    const int mofs  = (lane & 16) ? 8 : 0;
    const int n0 = w, n1 = w + 8;          // two column tiles per wave

    v8f c0 = {0.f,0.f,0.f,0.f,0.f,0.f,0.f,0.f};
    v8f c1 = {0.f,0.f,0.f,0.f,0.f,0.f,0.f,0.f};
    const size_t arow = (size_t)(mtile * 16 + (lane & 15)) * KEXT + ((lane & 16) ? 8 : 0);
    #pragma unroll 4
    for (int kc = 0; kc < KCN; ++kc) {
        const unsigned short* ar = x1e + arow + kc * 32;
        uint4 a0 = *(const uint4*)ar;
        uint4 a1 = *(const uint4*)(ar + 16);
        v16bf af = make_v16(a0, a1);
        const uint4* b0p = (const uint4*)(wext + (size_t)(((kc * 16 + n0) * 32 + lane) << 4));
        const uint4* b1p = (const uint4*)(wext + (size_t)(((kc * 16 + n1) * 32 + lane) << 4));
        v16bf bf0 = make_v16(b0p[0], b0p[1]);
        v16bf bf1 = make_v16(b1p[0], b1p[1]);
        c0 = __builtin_amdgcn_wmma_f32_16x16x32_bf16(false, af, false, bf0, (short)0, c0, false, false);
        c1 = __builtin_amdgcn_wmma_f32_16x16x32_bf16(false, af, false, bf1, (short)0, c1, false, false);
    }
    const int n = lane & 15;
    const int o0 = n0 * 16 + n, o1 = n1 * 16 + n;
    const float bias0 = pab[o0] + pnb[o0];
    const float bias1 = pab[o1] + pnb[o1];
    #pragma unroll
    for (int r = 0; r < 8; ++r) {
        int m = mtile * 16 + r + mofs;
        Y[(size_t)m * OUT_DIM + o0] = c0[r] + bias0;
        Y[(size_t)m * OUT_DIM + o1] = c1[r] + bias1;
    }
}

// ---------------- BatchNorm (training mode, biased var) + SELU ----------------
__global__ void bn_selu(const float* __restrict__ Y,
                        const float* __restrict__ gamma,
                        const float* __restrict__ beta,
                        float* __restrict__ out) {
    const int o = blockIdx.x;
    const int t = threadIdx.x;             // 128 threads, 4 rows each
    float v[4], s = 0.f, s2 = 0.f;
    #pragma unroll
    for (int i = 0; i < 4; ++i) {
        v[i] = Y[(size_t)(t + 128 * i) * OUT_DIM + o];
        s += v[i]; s2 += v[i] * v[i];
    }
    __shared__ float rs[128], rs2[128];
    rs[t] = s; rs2[t] = s2;
    __syncthreads();
    for (int off = 64; off; off >>= 1) {
        if (t < off) { rs[t] += rs[t + off]; rs2[t] += rs2[t + off]; }
        __syncthreads();
    }
    const float n   = (float)(B_ * T_);
    const float mu  = rs[0] / n;
    const float var = rs2[0] / n - mu * mu;
    const float rstd = rsqrtf(var + 1e-5f);
    const float g = gamma[o], be = beta[o];
    const float SC = 1.0507009873554805f, AL = 1.6732632423543772f;
    #pragma unroll
    for (int i = 0; i < 4; ++i) {
        float yn = (v[i] - mu) * rstd * g + be;
        float r  = yn > 0.f ? SC * yn : SC * AL * (__expf(yn) - 1.0f);
        out[(size_t)(t + 128 * i) * OUT_DIM + o] = r;
    }
}

// ---------------- launcher ----------------
extern "C" void kernel_launch(void* const* d_in, const int* in_sizes, int n_in,
                              void* d_out, int out_size, void* d_ws, size_t ws_size,
                              hipStream_t stream) {
    const float* x    = (const float*)d_in[0];
    const int*   bnd  = (const int*)  d_in[1];
    const float* apw  = (const float*)d_in[2];
    const float* apb  = (const float*)d_in[3];
    const float* awt  = (const float*)d_in[4];
    const float* paw  = (const float*)d_in[5];
    const float* pab  = (const float*)d_in[6];
    const float* pnw  = (const float*)d_in[7];
    const float* pnb  = (const float*)d_in[8];
    const float* bng  = (const float*)d_in[9];
    const float* bnb  = (const float*)d_in[10];
    float* out = (float*)d_out;

    char* ws = (char*)d_ws;
    unsigned short* wswz = (unsigned short*)(ws + OFF_WSWZ);
    unsigned short* xtp  = (unsigned short*)(ws + OFF_XT);
    unsigned short* wext = (unsigned short*)(ws + OFF_WEXT);
    unsigned short* aw2  = (unsigned short*)(ws + OFF_AW2);
    int*            S    = (int*)           (ws + OFF_S);
    unsigned short* x1e  = (unsigned short*)(ws + OFF_X1E);
    float*          Y    = (float*)         (ws + OFF_Y);

    prep_wswz  <<<131072 / 256, 256, 0, stream>>>(apw, wswz);
    prep_xt    <<<262144 / 256, 256, 0, stream>>>(x, xtp);
    prep_wext  <<<1179648 / 256, 256, 0, stream>>>(paw, pnw, wext);
    prep_aw2   <<<4096 / 256, 256, 0, stream>>>(awt, aw2);
    prep_x1tail<<<262144 / 256, 256, 0, stream>>>(x, x1e);
    prep_scan  <<<1, 32, 0, stream>>>(bnd, S);

    bam_main<<<B_ * T_, 256, 0, stream>>>(x, wswz, xtp, apb, aw2, S, x1e);

    gemm2<<<32, 256, 0, stream>>>(x1e, wext, pab, pnb, Y);

    bn_selu<<<OUT_DIM, 128, 0, stream>>>(Y, bng, bnb, out);
}
